// GAT_20916490732036
// MI455X (gfx1250) — matrix operand order repaired and checked by reading, not compiled, not run on hardware
//
#include <hip/hip_runtime.h>

// ---------------- constants from the reference ----------------
#define NN   50000
#define EE   400000
#define FS   256
#define HH   8
#define DD   64
#define HD   512      // H*D
#define BB   1024
#define NHD  512      // NH*DK

typedef __attribute__((ext_vector_type(16))) _Float16 v16h;
typedef __attribute__((ext_vector_type(8)))  float    v8f;

struct alignas(16) U128 { unsigned int x, y, z, w; };
union Frag16 { U128 q[2]; v16h v; };

// ---- fragment loaders (A row-major, BT = B pre-transposed) ----
__device__ __forceinline__ void load_a_frag(Frag16& a, const _Float16* arow, int k0)
{
    const U128* ap = reinterpret_cast<const U128*>(arow + k0);
    a.q[0] = ap[0];            // K = base+0..7
    a.q[1] = ap[2];            // K = base+16..23
}
__device__ __forceinline__ void load_b_frag(Frag16& b, const _Float16* BT,
                                            int n, int bofs, int K, int k0)
{
    const U128* bp = reinterpret_cast<const U128*>(BT + (size_t)n * K + k0 + bofs);
    b.q[0] = bp[0];
    b.q[1] = bp[2];
}

#define WMMA_STEP(AF, B0, B1, B2, B3)                                          \
    do {                                                                       \
        acc0 = __builtin_amdgcn_wmma_f32_16x16x32_f16(                         \
            false, (AF).v, false, (B0).v, (short)0, acc0, false, false);       \
        acc1 = __builtin_amdgcn_wmma_f32_16x16x32_f16(                         \
            false, (AF).v, false, (B1).v, (short)0, acc1, false, false);       \
        acc2 = __builtin_amdgcn_wmma_f32_16x16x32_f16(                         \
            false, (AF).v, false, (B2).v, (short)0, acc2, false, false);       \
        acc3 = __builtin_amdgcn_wmma_f32_16x16x32_f16(                         \
            false, (AF).v, false, (B3).v, (short)0, acc3, false, false);       \
    } while (0)

// =====================================================================
// WMMA GEMM:  C[M x Ncols] = A[M x K] (f16, row-major) * BT[Ncols x K]^T
// 2-deep software pipeline with NAMED register fragment sets (no runtime
// indexing -> everything stays in VGPRs). K must be a multiple of 64
// (true for all uses here: 256, 64, 512).
// Block = 256 threads = 8 waves; wave owns a 16 x 64 output strip.
// =====================================================================
__global__ __launch_bounds__(256)
void wmma_gemm_f16(const _Float16* __restrict__ A,
                   const _Float16* __restrict__ BT,
                   float* __restrict__ Cf, _Float16* __restrict__ Ch,
                   int M, int Ncols, int K, int storeHalf)
{
    const int lane = threadIdx.x & 31;
    const int wave = threadIdx.x >> 5;
    const int mtile = blockIdx.x * 8 + wave;
    const int mbase = mtile * 16;
    if (mbase >= M) return;
    const int n0  = blockIdx.y * 64;
    const int l15 = lane & 15;
    const int hi  = lane >> 4;                  // 0: lanes 0-15, 1: lanes 16-31

    // A fragment rows: M = mbase + l15 ; K runs {0..7,16..23} (+8 if hi)
    const _Float16* arow = A + (size_t)(mbase + l15) * K + (hi ? 8 : 0);
    // B fragment: column n, K runs {0..7,16..23} (+8 if hi)
    const int bofs = hi ? 8 : 0;
    const int nA = n0 + l15, nB = n0 + 16 + l15, nC = n0 + 32 + l15, nD = n0 + 48 + l15;

    v8f acc0 = {}, acc1 = {}, acc2 = {}, acc3 = {};
    Frag16 a0, b0, b1, b2, b3;      // even K-step fragments
    Frag16 a1, b4, b5, b6, b7;      // odd  K-step fragments

    // pipeline prologue: K-step 0 in flight
    load_a_frag(a0, arow, 0);
    load_b_frag(b0, BT, nA, bofs, K, 0);
    load_b_frag(b1, BT, nB, bofs, K, 0);
    load_b_frag(b2, BT, nC, bofs, K, 0);
    load_b_frag(b3, BT, nD, bofs, K, 0);

    for (int k0 = 0; k0 < K; k0 += 64) {
        // issue odd-step loads, then consume even-step fragments
        load_a_frag(a1, arow, k0 + 32);
        load_b_frag(b4, BT, nA, bofs, K, k0 + 32);
        load_b_frag(b5, BT, nB, bofs, K, k0 + 32);
        load_b_frag(b6, BT, nC, bofs, K, k0 + 32);
        load_b_frag(b7, BT, nD, bofs, K, k0 + 32);
        __builtin_prefetch(arow + k0 + 64, 0, 1);
        WMMA_STEP(a0, b0, b1, b2, b3);

        // issue next even-step loads, then consume odd-step fragments
        if (k0 + 64 < K) {
            load_a_frag(a0, arow, k0 + 64);
            load_b_frag(b0, BT, nA, bofs, K, k0 + 64);
            load_b_frag(b1, BT, nB, bofs, K, k0 + 64);
            load_b_frag(b2, BT, nC, bofs, K, k0 + 64);
            load_b_frag(b3, BT, nD, bofs, K, k0 + 64);
        }
        WMMA_STEP(a1, b4, b5, b6, b7);
    }

    // D layout: VGPR r -> row mbase + r (+8 if hi), col = n0 + j*16 + l15
    const int rowoff = mbase + (hi ? 8 : 0);
#pragma unroll
    for (int r = 0; r < 8; ++r) {
        size_t row = (size_t)(rowoff + r) * Ncols;
        if (storeHalf) {
            Ch[row + nA] = (_Float16)acc0[r];
            Ch[row + nB] = (_Float16)acc1[r];
            Ch[row + nC] = (_Float16)acc2[r];
            Ch[row + nD] = (_Float16)acc3[r];
        } else {
            Cf[row + nA] = acc0[r];
            Cf[row + nB] = acc1[r];
            Cf[row + nC] = acc2[r];
            Cf[row + nD] = acc3[r];
        }
    }
}

// ---------------- conversion / transpose helpers ----------------
__global__ void f32_to_f16_kernel(const float* __restrict__ in,
                                  _Float16* __restrict__ out, size_t n)
{
    size_t i = (size_t)blockIdx.x * blockDim.x + threadIdx.x;
    size_t stride = (size_t)gridDim.x * blockDim.x;
    for (; i < n; i += stride) out[i] = (_Float16)in[i];
}

// W[K x Ncols] (f32) -> WT[Ncols x K] (f16)
__global__ void transpose_to_f16_kernel(const float* __restrict__ W,
                                        _Float16* __restrict__ WT,
                                        int K, int Ncols)
{
    size_t i = (size_t)blockIdx.x * blockDim.x + threadIdx.x;
    size_t n = (size_t)K * Ncols;
    size_t stride = (size_t)gridDim.x * blockDim.x;
    for (; i < n; i += stride) {
        int k = (int)(i / Ncols), c = (int)(i % Ncols);
        WT[(size_t)c * K + k] = (_Float16)W[i];
    }
}

// ---------------- node -> batch-slot scatter ----------------
__global__ void n2b_scatter_kernel(const int* __restrict__ url,
                                   int* __restrict__ n2b)
{
    int b = blockIdx.x * blockDim.x + threadIdx.x;
    if (b < BB) n2b[url[b] - 1] = b;   // dup urls: last writer = canonical slot
}

// ---------------- el/er: per (node, head) dot products ----------------
__global__ __launch_bounds__(256)
void el_er_kernel(const _Float16* __restrict__ feat,
                  const float* __restrict__ attn_l,
                  const float* __restrict__ attn_r,
                  float* __restrict__ el, float* __restrict__ er)
{
    int n = blockIdx.x;
    int h = threadIdx.x >> 5;          // wave index = head
    int l = threadIdx.x & 31;
    const _Float16* fr = feat + (size_t)n * HD + h * DD + l * 2;
    float f0 = (float)fr[0], f1 = (float)fr[1];
    const float* al = attn_l + h * DD + l * 2;
    const float* ar = attn_r + h * DD + l * 2;
    float sl = f0 * al[0] + f1 * al[1];
    float sr = f0 * ar[0] + f1 * ar[1];
    for (int off = 16; off; off >>= 1) {
        sl += __shfl_down(sl, off, 32);
        sr += __shfl_down(sr, off, 32);
    }
    if (l == 0) { el[(size_t)n * HH + h] = sl; er[(size_t)n * HH + h] = sr; }
}

// ---------------- edge pass 1: softmax denominators (batch dsts only) ----
__global__ void edge_denom_kernel(const int* __restrict__ src,
                                  const int* __restrict__ dst,
                                  const int* __restrict__ n2b,
                                  const float* __restrict__ el,
                                  const float* __restrict__ er,
                                  float* __restrict__ denom)
{
    int e = blockIdx.x * blockDim.x + threadIdx.x;
    if (e >= EE) return;
    int d = dst[e];
    int b = n2b[d];
    if (b < 0) return;
    int s = src[e];
#pragma unroll
    for (int h = 0; h < HH; ++h) {
        float v = el[(size_t)s * HH + h] + er[(size_t)d * HH + h];
        v = v > 0.f ? v : 0.2f * v;                 // leaky_relu(0.2)
        atomicAdd(&denom[b * HH + h], expf(v));     // softmax shift-invariant
    }
}

// ---------------- edge pass 2: weighted message accumulation -------------
__global__ __launch_bounds__(256)
void edge_msg_kernel(const int* __restrict__ src,
                     const int* __restrict__ dst,
                     const int* __restrict__ n2b,
                     const float* __restrict__ el,
                     const float* __restrict__ er,
                     const float* __restrict__ denom,
                     const _Float16* __restrict__ feat,
                     float* __restrict__ rst)
{
    int wave = threadIdx.x >> 5, lane = threadIdx.x & 31;
    int e = blockIdx.x * 8 + wave;
    if (e >= EE) return;
    int d = dst[e];
    int b = n2b[d];
    if (b < 0) return;
    int s = src[e];
#pragma unroll
    for (int h = 0; h < HH; ++h) {
        float v = el[(size_t)s * HH + h] + er[(size_t)d * HH + h];
        v = v > 0.f ? v : 0.2f * v;
        float a = expf(v) / fmaxf(denom[b * HH + h], 1e-20f);
        int base = h * DD + lane * 2;
        float m0 = a * (float)feat[(size_t)s * HD + base];
        float m1 = a * (float)feat[(size_t)s * HD + base + 1];
        atomicAdd(&rst[(size_t)b * HD + base],     m0);
        atomicAdd(&rst[(size_t)b * HD + base + 1], m1);
    }
}

// ---------------- gather batch feature rows (f16) ----------------
__global__ void gather_feat_kernel(const _Float16* __restrict__ featB,
                                   const int* __restrict__ url,
                                   _Float16* __restrict__ out)
{
    size_t i = (size_t)blockIdx.x * blockDim.x + threadIdx.x;
    if (i >= (size_t)BB * FS) return;
    int b = (int)(i >> 8), j = (int)(i & (FS - 1));
    int node = url[b] - 1;
    out[i] = featB[(size_t)node * FS + j];
}

// ---------------- GAT epilogue: rst + res + bias, LayerNorm ----------------
__global__ __launch_bounds__(512)
void gat_finish_kernel(const int* __restrict__ url, const int* __restrict__ n2b,
                       const float* __restrict__ rst, const float* __restrict__ res,
                       const float* __restrict__ gat_bias,
                       const float* __restrict__ g, const float* __restrict__ bia,
                       float* __restrict__ x, _Float16* __restrict__ xh)
{
    int b = blockIdx.x, i = threadIdx.x, h = i >> 6, d = i & 63;
    int node = url[b] - 1;
    int cb = n2b[node];                       // canonical slot (handles dup urls)
    float val = rst[(size_t)cb * HD + i] + res[(size_t)b * HD + i] + gat_bias[i];
    __shared__ float sv[HD];
    __shared__ float smu[HH], srs[HH];
    sv[i] = val;
    __syncthreads();
    if (d == 0) {
        float su = 0.f, sq = 0.f;
        for (int t = 0; t < DD; ++t) { float v = sv[h * DD + t]; su += v; sq += v * v; }
        float mu = su * (1.f / DD);
        float var = sq * (1.f / DD) - mu * mu;
        smu[h] = mu; srs[h] = rsqrtf(var + 1e-6f);
    }
    __syncthreads();
    float y = (val - smu[h]) * srs[h] * g[d] + bia[d];
    x[(size_t)b * HD + i] = y;
    xh[(size_t)b * HD + i] = (_Float16)y;
}

// ---------------- tiny 8x8 multi-head attention per (b, nh) ----------------
__global__ __launch_bounds__(64)
void mha_attn_kernel(const float* __restrict__ q, const float* __restrict__ k,
                     const float* __restrict__ v,
                     float* __restrict__ ao, _Float16* __restrict__ aoh)
{
    int b = blockIdx.x, nh = blockIdx.y, t = threadIdx.x;
    __shared__ float qs[HH][DD], ks[HH][DD], vs[HH][DD];
    __shared__ float sc[HH][HH], aw[HH][HH];
    for (int s = 0; s < HH; ++s) {
        size_t idx = ((size_t)(b * HH + s)) * NHD + nh * DD + t;
        qs[s][t] = q[idx]; ks[s][t] = k[idx]; vs[s][t] = v[idx];
    }
    __syncthreads();
    {
        int qi = t >> 3, ki = t & 7;
        float s = 0.f;
        for (int dd = 0; dd < DD; ++dd) s += qs[qi][dd] * ks[ki][dd];
        sc[qi][ki] = s * 0.125f;              // 1/sqrt(64)
    }
    __syncthreads();
    if (t < HH) {
        float mx = -1e30f;
        for (int j = 0; j < HH; ++j) mx = fmaxf(mx, sc[t][j]);
        float sum = 0.f, ex[HH];
        for (int j = 0; j < HH; ++j) { ex[j] = expf(sc[t][j] - mx); sum += ex[j]; }
        float inv = 1.f / sum;
        for (int j = 0; j < HH; ++j) aw[t][j] = ex[j] * inv;
    }
    __syncthreads();
    for (int qi = 0; qi < HH; ++qi) {
        float o = 0.f;
        for (int j = 0; j < HH; ++j) o += aw[qi][j] * vs[j][t];
        size_t idx = ((size_t)(b * HH + qi)) * NHD + nh * DD + t;
        ao[idx] = o; aoh[idx] = (_Float16)o;
    }
}

// ---------------- MHA epilogue: +fc_b, +residual, LayerNorm ----------------
__global__ __launch_bounds__(512)
void mha_finish_kernel(const float* __restrict__ o2, const float* __restrict__ fc_b,
                       const float* __restrict__ xres,
                       const float* __restrict__ g, const float* __restrict__ bia,
                       float* __restrict__ ofin)
{
    int b = blockIdx.x, i = threadIdx.x, h = i >> 6, d = i & 63;
    float val = o2[(size_t)b * HD + i] + fc_b[d] + xres[(size_t)b * HD + i];
    __shared__ float sv[HD];
    __shared__ float smu[HH], srs[HH];
    sv[i] = val;
    __syncthreads();
    if (d == 0) {
        float su = 0.f, sq = 0.f;
        for (int t = 0; t < DD; ++t) { float v = sv[h * DD + t]; su += v; sq += v * v; }
        float mu = su * (1.f / DD);
        float var = sq * (1.f / DD) - mu * mu;
        smu[h] = mu; srs[h] = rsqrtf(var + 1e-6f);
    }
    __syncthreads();
    ofin[(size_t)b * HD + i] = (val - smu[h]) * srs[h] * g[d] + bia[d];
}

// ---------------- pool over heads + 2-way classifier ----------------
__global__ __launch_bounds__(64)
void pool_cls_kernel(const float* __restrict__ ofin,
                     const float* __restrict__ out_w, const float* __restrict__ out_b,
                     float* __restrict__ out)
{
    int b = blockIdx.x, d = threadIdx.x;
    float p = 0.f;
#pragma unroll
    for (int h = 0; h < HH; ++h) p += ofin[(size_t)b * HD + h * DD + d];
    __shared__ float sp[DD];
    sp[d] = p;
    __syncthreads();
    if (d < 2) {
        float acc = out_b[d];
        for (int j = 0; j < DD; ++j) acc += sp[j] * out_w[j * 2 + d];
        out[b * 2 + d] = acc;
    }
}

// =====================================================================
extern "C" void kernel_launch(void* const* d_in, const int* in_sizes, int n_in,
                              void* d_out, int out_size, void* d_ws, size_t ws_size,
                              hipStream_t stream)
{
    const float* features = (const float*)d_in[0];
    const int*   src      = (const int*)d_in[1];
    const int*   dst      = (const int*)d_in[2];
    const int*   url      = (const int*)d_in[3];
    const float* gat_fc_w = (const float*)d_in[4];
    const float* attn_l   = (const float*)d_in[5];
    const float* attn_r   = (const float*)d_in[6];
    const float* gat_res_w= (const float*)d_in[7];
    const float* gat_bias = (const float*)d_in[8];
    const float* ln_g     = (const float*)d_in[9];
    const float* ln_b     = (const float*)d_in[10];
    const float* wq       = (const float*)d_in[11];
    const float* wk       = (const float*)d_in[12];
    const float* wv       = (const float*)d_in[13];
    const float* fc_w     = (const float*)d_in[14];
    const float* fc_b     = (const float*)d_in[15];
    const float* mha_ln_g = (const float*)d_in[16];
    const float* mha_ln_b = (const float*)d_in[17];
    const float* out_w    = (const float*)d_in[18];
    const float* out_b    = (const float*)d_in[19];
    float* out = (float*)d_out;

    // -------- workspace carve-up (256B aligned) --------
    char* ws = (char*)d_ws;
    size_t off = 0;
    auto alloc = [&](size_t bytes) -> char* {
        char* p = ws + off;
        off = (off + bytes + 255) & ~(size_t)255;
        return p;
    };
    _Float16* featB  = (_Float16*)alloc((size_t)NN * FS * 2);   // features f16
    _Float16* feat_h = (_Float16*)alloc((size_t)NN * HD * 2);   // feat (post fc) f16
    _Float16* WfcT   = (_Float16*)alloc((size_t)HD * FS * 2);
    _Float16* WresT  = (_Float16*)alloc((size_t)HD * FS * 2);
    _Float16* wqT    = (_Float16*)alloc((size_t)NHD * DD * 2);
    _Float16* wkT    = (_Float16*)alloc((size_t)NHD * DD * 2);
    _Float16* wvT    = (_Float16*)alloc((size_t)NHD * DD * 2);
    _Float16* fcT    = (_Float16*)alloc((size_t)DD * NHD * 2);
    float*    el     = (float*)alloc((size_t)NN * HH * 4);
    float*    er     = (float*)alloc((size_t)NN * HH * 4);
    int*      n2b    = (int*)alloc((size_t)NN * 4);
    float*    denom  = (float*)alloc((size_t)BB * HH * 4);
    float*    rst    = (float*)alloc((size_t)BB * HD * 4);
    _Float16* featBb = (_Float16*)alloc((size_t)BB * FS * 2);
    float*    res    = (float*)alloc((size_t)BB * HD * 4);
    float*    x      = (float*)alloc((size_t)BB * HD * 4);
    _Float16* xh     = (_Float16*)alloc((size_t)BB * HD * 2);
    float*    qf     = (float*)alloc((size_t)BB * HH * NHD * 4);
    float*    kf     = (float*)alloc((size_t)BB * HH * NHD * 4);
    float*    vf     = (float*)alloc((size_t)BB * HH * NHD * 4);
    float*    ao     = (float*)alloc((size_t)BB * HH * NHD * 4);
    _Float16* aoh    = (_Float16*)alloc((size_t)BB * HH * NHD * 2);
    float*    o2     = (float*)alloc((size_t)BB * HD * 4);
    float*    ofin   = (float*)alloc((size_t)BB * HD * 4);
    (void)ws_size; (void)in_sizes; (void)n_in; (void)out_size;

    // -------- precision conversion + weight transposes --------
    {
        size_t n = (size_t)NN * FS;
        f32_to_f16_kernel<<<(int)((n + 255) / 256), 256, 0, stream>>>(features, featB, n);
    }
    transpose_to_f16_kernel<<<512, 256, 0, stream>>>(gat_fc_w,  WfcT,  FS,  HD);
    transpose_to_f16_kernel<<<512, 256, 0, stream>>>(gat_res_w, WresT, FS,  HD);
    transpose_to_f16_kernel<<<128, 256, 0, stream>>>(wq,   wqT, DD,  NHD);
    transpose_to_f16_kernel<<<128, 256, 0, stream>>>(wk,   wkT, DD,  NHD);
    transpose_to_f16_kernel<<<128, 256, 0, stream>>>(wv,   wvT, DD,  NHD);
    transpose_to_f16_kernel<<<128, 256, 0, stream>>>(fc_w, fcT, NHD, DD);

    // -------- zero/flag scratch (graph-capture safe) --------
    hipMemsetAsync(n2b,   0xFF, (size_t)NN * 4, stream);       // -1
    hipMemsetAsync(denom, 0,    (size_t)BB * HH * 4, stream);
    hipMemsetAsync(rst,   0,    (size_t)BB * HD * 4, stream);
    n2b_scatter_kernel<<<(BB + 255) / 256, 256, 0, stream>>>(url, n2b);

    // -------- big GEMM: feat = features @ gat_fc_w  (f16 out) --------
    wmma_gemm_f16<<<dim3((NN + 127) / 128, HD / 64), 256, 0, stream>>>(
        featB, WfcT, nullptr, feat_h, NN, HD, FS, 1);

    // -------- per-(node,head) attention logits --------
    el_er_kernel<<<NN, 256, 0, stream>>>(feat_h, attn_l, attn_r, el, er);

    // -------- batch residual GEMM --------
    {
        size_t n = (size_t)BB * FS;
        gather_feat_kernel<<<(int)((n + 255) / 256), 256, 0, stream>>>(featB, url, featBb);
    }
    wmma_gemm_f16<<<dim3(BB / 128, HD / 64), 256, 0, stream>>>(
        featBb, WresT, res, nullptr, BB, HD, FS, 0);

    // -------- edge softmax + aggregation (batch destinations only) --------
    edge_denom_kernel<<<(EE + 255) / 256, 256, 0, stream>>>(src, dst, n2b, el, er, denom);
    edge_msg_kernel<<<(EE + 7) / 8, 256, 0, stream>>>(src, dst, n2b, el, er, denom, feat_h, rst);

    // -------- GAT epilogue + LayerNorm --------
    gat_finish_kernel<<<BB, 512, 0, stream>>>(url, n2b, rst, res, gat_bias, ln_g, ln_b, x, xh);

    // -------- QKV GEMMs: [B*H, 64] @ [64, 512] --------
    wmma_gemm_f16<<<dim3((BB * HH) / 128, NHD / 64), 256, 0, stream>>>(
        xh, wqT, qf, nullptr, BB * HH, NHD, DD, 0);
    wmma_gemm_f16<<<dim3((BB * HH) / 128, NHD / 64), 256, 0, stream>>>(
        xh, wkT, kf, nullptr, BB * HH, NHD, DD, 0);
    wmma_gemm_f16<<<dim3((BB * HH) / 128, NHD / 64), 256, 0, stream>>>(
        xh, wvT, vf, nullptr, BB * HH, NHD, DD, 0);

    // -------- 8x8 attention per (b, nh) --------
    mha_attn_kernel<<<dim3(BB, HH), 64, 0, stream>>>(qf, kf, vf, ao, aoh);

    // -------- output projection GEMM: [B*H, 512] @ [512, 64] --------
    wmma_gemm_f16<<<dim3((BB * HH) / 128, DD / 64), 256, 0, stream>>>(
        aoh, fcT, o2, nullptr, BB * HH, DD, NHD, 0);

    // -------- MHA epilogue + LayerNorm --------
    mha_finish_kernel<<<BB, 512, 0, stream>>>(o2, fc_b, x, mha_ln_g, mha_ln_b, ofin);

    // -------- pool + classify --------
    pool_cls_kernel<<<BB, 64, 0, stream>>>(ofin, out_w, out_b, out);
}
